// PaiNN_21835613733024
// MI455X (gfx1250) — matrix operand order repaired and checked by reading, not compiled, not run on hardware
//
#include <hip/hip_runtime.h>
#include <hip/hip_bf16.h>
#include <math.h>

#define BB 4
#define NN 512
#define JJ 48
#define FEAT 128
#define NRBF 20
#define NLAYER 3
#define BN (BB*NN)
#define PAIR (BB*NN*JJ)
#define PI_F 3.14159265358979f
#define CUTOFF_F 5.0f

typedef __attribute__((ext_vector_type(16))) _Float16 v16h;
typedef __attribute__((ext_vector_type(8)))  float    v8f;

// ---------------------------------------------------------------------------
// Pack W (f32, element [k][c] at W[k*ld + coloff + c]; rows >= srcK are zero)
// into f16 WMMA B-fragment order:
//   flat[(((kb/32)*(N/16) + nt)*32 + lane)*16 + i]
//     = W[kb + (lane>>4)*16 + i, nt*16 + (lane&15)]
// so GEMM fetches one contiguous 32B v16h per (tile, lane).
// ---------------------------------------------------------------------------
__global__ void pack_w_kernel(const float* __restrict__ W,
                              _Float16* __restrict__ Wp,
                              int K, int N, int srcK, int ld, int coloff)
{
  int idx = blockIdx.x * blockDim.x + threadIdx.x;
  if (idx >= K * N) return;
  int i    = idx & 15;
  int lane = (idx >> 4) & 31;
  int rest = idx >> 9;
  int nNt  = N >> 4;
  int nt   = rest % nNt;
  int kb   = rest / nNt;
  int n = nt * 16 + (lane & 15);
  int k = kb * 32 + (lane >> 4) * 16 + i;
  Wp[idx] = (k < srcK) ? (_Float16)W[(size_t)k * ld + coloff + n]
                       : (_Float16)0.f;
}

// ---------------------------------------------------------------------------
// WMMA GEMM: OUT[M,N] = (silu?)((A[M,K] @ W + bias) * (rowscale?)), W packed.
// One wave per 16(M) x 64(N) strip; per K-chunk all 4 packed-B fragments are
// loaded into distinct registers before the 4 v_wmma so load latency overlaps
// all four matrix ops. K compile-time. Optional f16 output.
// Grid: (M/16) * (N/64) blocks of 32 threads.
// ---------------------------------------------------------------------------
template <int K, int DOSILU, int ROWSCALE, int OUTF16>
__global__ __launch_bounds__(32) void gemm_wmma(
    const float* __restrict__ A, const _Float16* __restrict__ Wp,
    const float* __restrict__ bias, float* __restrict__ OUT, int N,
    const float* __restrict__ rs)
{
  const int ngrp = N >> 6;                 // N/64 strips
  const int mt = blockIdx.x / ngrp;
  const int ng = blockIdx.x % ngrp;
  const int lane = threadIdx.x;
  const int lm = lane & 15;
  const int hi = lane >> 4;
  const int nNt = N >> 4;

  const int m = mt * 16 + lm;
  const float* __restrict__ arow = A + (size_t)m * K;

  v8f acc[4] = {v8f{}, v8f{}, v8f{}, v8f{}};

#pragma unroll
  for (int kb = 0; kb < K; kb += 32) {
    // ---- A fragment (contiguous, matches packed B): lanes 0-15 K kb+0..15;
    //      lanes 16-31 K kb+16..31.
    const int abase = kb + hi * 16;
    const float4 a0 = *reinterpret_cast<const float4*>(arow + abase);
    const float4 a1 = *reinterpret_cast<const float4*>(arow + abase + 4);
    const float4 a2 = *reinterpret_cast<const float4*>(arow + abase + 8);
    const float4 a3 = *reinterpret_cast<const float4*>(arow + abase + 12);
    v16h a;
    a[0] = (_Float16)a0.x;  a[1] = (_Float16)a0.y;
    a[2] = (_Float16)a0.z;  a[3] = (_Float16)a0.w;
    a[4] = (_Float16)a1.x;  a[5] = (_Float16)a1.y;
    a[6] = (_Float16)a1.z;  a[7] = (_Float16)a1.w;
    a[8] = (_Float16)a2.x;  a[9] = (_Float16)a2.y;
    a[10] = (_Float16)a2.z; a[11] = (_Float16)a2.w;
    a[12] = (_Float16)a3.x; a[13] = (_Float16)a3.y;
    a[14] = (_Float16)a3.z; a[15] = (_Float16)a3.w;

    // ---- All 4 packed-B fragments first (distinct regs), then 4 WMMAs.
    const _Float16* __restrict__ fb =
        Wp + (((size_t)(kb >> 5) * nNt + (size_t)ng * 4) * 32 + lane) * 16;
    const v16h b0 = *reinterpret_cast<const v16h*>(fb);
    const v16h b1 = *reinterpret_cast<const v16h*>(fb + 512);
    const v16h b2 = *reinterpret_cast<const v16h*>(fb + 1024);
    const v16h b3 = *reinterpret_cast<const v16h*>(fb + 1536);
    acc[0] = __builtin_amdgcn_wmma_f32_16x16x32_f16(false, a, false, b0,
                                                    (short)0, acc[0], false, false);
    acc[1] = __builtin_amdgcn_wmma_f32_16x16x32_f16(false, a, false, b1,
                                                    (short)0, acc[1], false, false);
    acc[2] = __builtin_amdgcn_wmma_f32_16x16x32_f16(false, a, false, b2,
                                                    (short)0, acc[2], false, false);
    acc[3] = __builtin_amdgcn_wmma_f32_16x16x32_f16(false, a, false, b3,
                                                    (short)0, acc[3], false, false);
  }

  // ---- D layout: VGPR r -> row mt*16 + hi*8 + r, column n.
  const int rowbase = mt * 16 + hi * 8;
#pragma unroll
  for (int t = 0; t < 4; ++t) {
    const int n = ng * 64 + t * 16 + lm;
    const float bv = bias ? bias[n] : 0.f;
#pragma unroll
    for (int r = 0; r < 8; ++r) {
      float v = acc[t][r] + bv;
      if (ROWSCALE) v *= rs[rowbase + r];
      if (DOSILU) v = v / (1.f + expf(-v));
      if (OUTF16)
        ((_Float16*)OUT)[(size_t)(rowbase + r) * N + n] = (_Float16)v;
      else
        OUT[(size_t)(rowbase + r) * N + n] = v;
    }
  }
}

// ---------------------------------------------------------------------------
// Zero-pad node_features [BN,100] -> [BN,128] for the guard-free K=128 path.
// ---------------------------------------------------------------------------
__global__ void pad_node_kernel(const float* __restrict__ src,
                                float* __restrict__ dst)
{
  int i = blockIdx.x * blockDim.x + threadIdx.x;
  if (i >= BN * 128) return;
  int m = i >> 7, k = i & 127;
  dst[i] = (k < 100) ? src[(size_t)m * 100 + k] : 0.f;
}

// ---------------------------------------------------------------------------
// Geometry: distances, unit directions, Bessel RBF phi (padded to 32 cols so
// the filter GEMM takes the guard-free K=32 path), cosine cutoff.
// ---------------------------------------------------------------------------
__global__ void geom_kernel(const float* __restrict__ pos,
                            const int* __restrict__ nbr,
                            const float* __restrict__ mask,
                            float* __restrict__ phi32, float* __restrict__ fcut,
                            float* __restrict__ dirv)
{
  int idx = blockIdx.x * blockDim.x + threadIdx.x;
  if (idx >= PAIR) return;
  int b   = idx / (NN * JJ);
  int rem = idx % (NN * JJ);
  int n   = rem / JJ;
  int nb  = nbr[idx];
  const float* pi = pos + ((size_t)b * NN + n) * 3;
  const float* pj = pos + ((size_t)b * NN + nb) * 3;
  float vx = pj[0] - pi[0], vy = pj[1] - pi[1], vz = pj[2] - pi[2];
  float raw = sqrtf(vx * vx + vy * vy + vz * vz);
  float mk = mask[idx];
  float d  = raw * mk;
  float rinv = (raw == 0.f) ? 1.f : (1.f / raw);
  dirv[(size_t)idx * 3 + 0] = vx * rinv;
  dirv[(size_t)idx * 3 + 1] = vy * rinv;
  dirv[(size_t)idx * 3 + 2] = vz * rinv;
  float dinv = (d == 0.f) ? 1.f : (1.f / d);
#pragma unroll
  for (int k = 0; k < NRBF; ++k) {
    float fr = (float)(k + 1) * (PI_F / CUTOFF_F);
    phi32[(size_t)idx * 32 + k] = sinf(d * fr) * dinv;
  }
#pragma unroll
  for (int k = NRBF; k < 32; ++k) phi32[(size_t)idx * 32 + k] = 0.f;
  float fc = 0.5f * (cosf(d * (PI_F / CUTOFF_F)) + 1.f) *
             ((d < CUTOFF_F) ? 1.f : 0.f) * mk;
  fcut[idx] = fc;
}

// ---------------------------------------------------------------------------
// Message pass, filters precomputed (f16, fcut/bias already folded by the
// WMMA filter GEMM). Gathers h_j / v_j (L2-resident), reduces over J.
// vectors double-buffered (vin read at pre-update values).
// ---------------------------------------------------------------------------
__global__ __launch_bounds__(128) void message_pre_kernel(
    float* __restrict__ scalars, const float* __restrict__ vin,
    float* __restrict__ vout, const float* __restrict__ h,
    const int* __restrict__ nbr, const _Float16* __restrict__ filt,
    const float* __restrict__ dirv)
{
  __shared__ float sdir[JJ * 3];
  __shared__ int   snb[JJ];

  int bn  = blockIdx.x;
  int b   = bn / NN;
  int tid = threadIdx.x;
  size_t p0 = (size_t)bn * JJ;
  for (int i = tid; i < JJ * 3; i += 128) sdir[i] = dirv[p0 * 3 + i];
  if (tid < JJ) snb[tid] = nbr[p0 + tid];
  __syncthreads();

  int f = tid;
  float ds = 0.f, dv0 = 0.f, dv1 = 0.f, dv2 = 0.f;
  for (int j = 0; j < JJ; ++j) {
    int nb = snb[j];
    const float* hb = h   + ((size_t)b * NN + nb) * 384;
    const float* vb = vin + ((size_t)b * NN + nb) * 384;
    const _Float16* fr = filt + (p0 + j) * 384;
    float w0 = (float)fr[f];
    float w1 = (float)fr[128 + f];
    float w2 = (float)fr[256 + f];
    ds += w0 * hb[f];
    float dvR = w1 * hb[128 + f];
    float dvv = w2 * hb[256 + f];
    dv0 += dvR * sdir[j * 3 + 0] + dvv * vb[f];
    dv1 += dvR * sdir[j * 3 + 1] + dvv * vb[128 + f];
    dv2 += dvR * sdir[j * 3 + 2] + dvv * vb[256 + f];
  }
  scalars[(size_t)bn * 128 + f] += ds;
  size_t vo = (size_t)bn * 384;
  vout[vo + f]       = vin[vo + f]       + dv0;
  vout[vo + 128 + f] = vin[vo + 128 + f] + dv1;
  vout[vo + 256 + f] = vin[vo + 256 + f] + dv2;
}

// ---------------------------------------------------------------------------
// Fallback message pass (if ws too small for the filter tensor): filters
// computed on the fly from phi32 with the LDS-cached 20x384 weight slice.
// ---------------------------------------------------------------------------
__global__ __launch_bounds__(128) void message_kernel(
    float* __restrict__ scalars, const float* __restrict__ vin,
    float* __restrict__ vout, const float* __restrict__ h,
    const int* __restrict__ nbr, const float* __restrict__ phi32,
    const float* __restrict__ fcut, const float* __restrict__ dirv,
    const float* __restrict__ filtW, const float* __restrict__ filtB, int off)
{
  __shared__ float sW[NRBF * 384];
  __shared__ float sB[384];
  __shared__ float sphi[JJ * NRBF];
  __shared__ float sdir[JJ * 3];
  __shared__ float sfc[JJ];
  __shared__ int   snb[JJ];

  int bn  = blockIdx.x;
  int b   = bn / NN;
  int tid = threadIdx.x;

  for (int i = tid; i < NRBF * 384; i += 128) {
    int k = i / 384, c2 = i % 384;
    sW[i] = filtW[(size_t)k * (NLAYER * 384) + off + c2];
  }
  for (int i = tid; i < 384; i += 128) sB[i] = filtB[off + i];
  size_t p0 = (size_t)bn * JJ;
  for (int i = tid; i < JJ * NRBF; i += 128) {
    int j = i / NRBF, k = i % NRBF;
    sphi[i] = phi32[(p0 + j) * 32 + k];
  }
  for (int i = tid; i < JJ * 3; i += 128) sdir[i] = dirv[p0 * 3 + i];
  if (tid < JJ) { sfc[tid] = fcut[p0 + tid]; snb[tid] = nbr[p0 + tid]; }
  __syncthreads();

  int f = tid;
  float ds = 0.f, dv0 = 0.f, dv1 = 0.f, dv2 = 0.f;
  for (int j = 0; j < JJ; ++j) {
    int nb = snb[j];
    const float* hb = h   + ((size_t)b * NN + nb) * 384;
    const float* vb = vin + ((size_t)b * NN + nb) * 384;
    float w0 = sB[f], w1 = sB[128 + f], w2 = sB[256 + f];
    const float* ph = sphi + j * NRBF;
#pragma unroll
    for (int k = 0; k < NRBF; ++k) {
      float p = ph[k];
      w0 += p * sW[k * 384 + f];
      w1 += p * sW[k * 384 + 128 + f];
      w2 += p * sW[k * 384 + 256 + f];
    }
    float fc = sfc[j];
    w0 *= fc; w1 *= fc; w2 *= fc;
    ds += w0 * hb[f];
    float dvR = w1 * hb[128 + f];
    float dvv = w2 * hb[256 + f];
    dv0 += dvR * sdir[j * 3 + 0] + dvv * vb[f];
    dv1 += dvR * sdir[j * 3 + 1] + dvv * vb[128 + f];
    dv2 += dvR * sdir[j * 3 + 2] + dvv * vb[256 + f];
  }
  scalars[(size_t)bn * 128 + f] += ds;
  size_t vo = (size_t)bn * 384;
  vout[vo + f]       = vin[vo + f]       + dv0;
  vout[vo + 128 + f] = vin[vo + 128 + f] + dv1;
  vout[vo + 256 + f] = vin[vo + 256 + f] + dv2;
}

// ---------------------------------------------------------------------------
// Post-mu: Vn = ||V||, ctx = [scalars, Vn], VU = sum_d V*U
// ---------------------------------------------------------------------------
__global__ __launch_bounds__(128) void post_kernel(
    const float* __restrict__ mu, const float* __restrict__ scalars,
    float* __restrict__ ctx, float* __restrict__ vu)
{
  int bn = blockIdx.x, f = threadIdx.x;
  size_t base = (size_t)bn * 3 * 256;
  float v0 = mu[base + f],       v1 = mu[base + 256 + f], v2 = mu[base + 512 + f];
  float u0 = mu[base + 128 + f], u1 = mu[base + 384 + f], u2 = mu[base + 640 + f];
  float vn = sqrtf(v0 * v0 + v1 * v1 + v2 * v2);
  ctx[(size_t)bn * 256 + f]       = scalars[(size_t)bn * 128 + f];
  ctx[(size_t)bn * 256 + 128 + f] = vn;
  vu[(size_t)bn * 128 + f] = v0 * u0 + v1 * u1 + v2 * u2;
}

// ---------------------------------------------------------------------------
// Intra-atom update: scalars += ds2 + dsv*VU ; vectors += dv * U
// ---------------------------------------------------------------------------
__global__ __launch_bounds__(128) void update_kernel(
    const float* __restrict__ h2, const float* __restrict__ mu,
    const float* __restrict__ vu, float* __restrict__ scalars,
    float* __restrict__ vec)
{
  int bn = blockIdx.x, f = threadIdx.x;
  size_t hb = (size_t)bn * 384;
  float ds2 = h2[hb + f], dvf = h2[hb + 128 + f], dsv = h2[hb + 256 + f];
  scalars[(size_t)bn * 128 + f] += ds2 + dsv * vu[(size_t)bn * 128 + f];
  size_t mb = (size_t)bn * 3 * 256;
  size_t vb = (size_t)bn * 384;
  vec[vb + f]       += dvf * mu[mb + 128 + f];
  vec[vb + 128 + f] += dvf * mu[mb + 384 + f];
  vec[vb + 256 + f] += dvf * mu[mb + 640 + f];
}

__global__ void zero_kernel(float* __restrict__ p, int n)
{
  int i = blockIdx.x * blockDim.x + threadIdx.x;
  if (i < n) p[i] = 0.f;
}

// ---------------------------------------------------------------------------
extern "C" void kernel_launch(void* const* d_in, const int* in_sizes, int n_in,
                              void* d_out, int out_size, void* d_ws,
                              size_t ws_size, hipStream_t stream)
{
  (void)in_sizes; (void)n_in; (void)out_size;
  const float* node  = (const float*)d_in[0];
  const float* pos   = (const float*)d_in[1];
  const int*   nbr   = (const int*)d_in[2];
  const float* nmask = (const float*)d_in[3];
  /* d_in[4] atom_mask unused by reference */
  const float* embW  = (const float*)d_in[5];
  const float* embB  = (const float*)d_in[6];
  const float* filtW = (const float*)d_in[7];
  const float* filtB = (const float*)d_in[8];
  const float* iW1   = (const float*)d_in[9];
  const float* ib1   = (const float*)d_in[10];
  const float* iW2   = (const float*)d_in[11];
  const float* ib2   = (const float*)d_in[12];
  const float* aW1   = (const float*)d_in[13];
  const float* ab1   = (const float*)d_in[14];
  const float* aW2   = (const float*)d_in[15];
  const float* ab2   = (const float*)d_in[16];
  const float* muW   = (const float*)d_in[17];

  float* scalars = (float*)d_out;            // [BN,128] lives in d_out

  float* ws    = (float*)d_ws;
  float* vecA  = ws; ws += (size_t)BN * 384;
  float* vecB  = ws; ws += (size_t)BN * 384;
  float* phi32 = ws; ws += (size_t)PAIR * 32;
  float* fcut  = ws; ws += (size_t)PAIR;
  float* dirv  = ws; ws += (size_t)PAIR * 3;
  float* h1    = ws; ws += (size_t)BN * 128;
  float* hbuf  = ws; ws += (size_t)BN * 384;
  float* mu    = ws; ws += (size_t)BN * 3 * 256;
  float* ctx   = ws; ws += (size_t)BN * 256;
  float* vu    = ws; ws += (size_t)BN * 128;
  float* Apad  = ws; ws += (size_t)BN * 128;  // node_features zero-padded

  _Float16* hw   = (_Float16*)ws;             // packed f16 weights
  _Float16* pEmb = hw; hw += (size_t)128 * 128;
  _Float16* piW1 = hw; hw += (size_t)NLAYER * 128 * 128;
  _Float16* piW2 = hw; hw += (size_t)NLAYER * 128 * 384;
  _Float16* pmuW = hw; hw += (size_t)NLAYER * 128 * 256;
  _Float16* paW1 = hw; hw += (size_t)NLAYER * 256 * 128;
  _Float16* paW2 = hw; hw += (size_t)NLAYER * 128 * 384;
  _Float16* pfW  = hw; hw += (size_t)NLAYER * 32 * 384;
  _Float16* filt = hw;                        // optional [PAIR,384] f16
  const size_t needed =
      (size_t)((char*)(filt + (size_t)PAIR * 384) - (char*)d_ws);
  const bool usePre = (ws_size >= needed);    // fixed per process -> graph-safe

  // ---- pack all weights into WMMA B-fragment f16 (tiny one-time pass) ----
  pack_w_kernel<<<(128 * 128 + 255) / 256, 256, 0, stream>>>(
      embW, pEmb, 128, 128, 100, 128, 0);
  for (int i = 0; i < NLAYER; ++i) {
    pack_w_kernel<<<(128 * 128 + 255) / 256, 256, 0, stream>>>(
        iW1 + (size_t)i * 128 * 128, piW1 + (size_t)i * 128 * 128,
        128, 128, 128, 128, 0);
    pack_w_kernel<<<(128 * 384 + 255) / 256, 256, 0, stream>>>(
        iW2 + (size_t)i * 128 * 384, piW2 + (size_t)i * 128 * 384,
        128, 384, 128, 384, 0);
    pack_w_kernel<<<(128 * 256 + 255) / 256, 256, 0, stream>>>(
        muW + (size_t)i * 128 * 256, pmuW + (size_t)i * 128 * 256,
        128, 256, 128, 256, 0);
    pack_w_kernel<<<(256 * 128 + 255) / 256, 256, 0, stream>>>(
        aW1 + (size_t)i * 256 * 128, paW1 + (size_t)i * 256 * 128,
        256, 128, 256, 128, 0);
    pack_w_kernel<<<(128 * 384 + 255) / 256, 256, 0, stream>>>(
        aW2 + (size_t)i * 128 * 384, paW2 + (size_t)i * 128 * 384,
        128, 384, 128, 384, 0);
    // filter slice: [20,1152] cols i*384..i*384+383, K padded 20->32
    pack_w_kernel<<<(32 * 384 + 255) / 256, 256, 0, stream>>>(
        filtW, pfW + (size_t)i * 32 * 384, 32, 384, NRBF, NLAYER * 384,
        i * 384);
  }

  zero_kernel<<<(BN * 384 + 255) / 256, 256, 0, stream>>>(vecA, BN * 384);
  pad_node_kernel<<<(BN * 128 + 255) / 256, 256, 0, stream>>>(node, Apad);
  geom_kernel<<<(PAIR + 255) / 256, 256, 0, stream>>>(pos, nbr, nmask, phi32,
                                                      fcut, dirv);
  // embedding: scalars = Apad @ embW + embB   [2048,128]x[128,128]
  gemm_wmma<128, 0, 0, 0><<<(BN / 16) * (128 / 64), 32, 0, stream>>>(
      Apad, pEmb, embB, scalars, 128, nullptr);

  float* cur = vecA; float* alt = vecB;
  for (int i = 0; i < NLAYER; ++i) {
    // h1 = silu(scalars @ iW1 + ib1)          [2048,128]x[128,128]
    gemm_wmma<128, 1, 0, 0><<<(BN / 16) * (128 / 64), 32, 0, stream>>>(
        scalars, piW1 + (size_t)i * 128 * 128, ib1 + i * 128, h1, 128, nullptr);
    // h = h1 @ iW2 + ib2                      [2048,128]x[128,384]
    gemm_wmma<128, 0, 0, 0><<<(BN / 16) * (384 / 64), 32, 0, stream>>>(
        h1, piW2 + (size_t)i * 128 * 384, ib2 + i * 384, hbuf, 384, nullptr);
    if (usePre) {
      // filt = (phi32 @ Wslice + b) * fcut    [98304,32]x[32,384] -> f16
      gemm_wmma<32, 0, 1, 1><<<(PAIR / 16) * (384 / 64), 32, 0, stream>>>(
          phi32, pfW + (size_t)i * 32 * 384, filtB + i * 384, (float*)filt,
          384, fcut);
      message_pre_kernel<<<BN, 128, 0, stream>>>(scalars, cur, alt, hbuf, nbr,
                                                 filt, dirv);
    } else {
      message_kernel<<<BN, 128, 0, stream>>>(scalars, cur, alt, hbuf, nbr,
                                             phi32, fcut, dirv, filtW, filtB,
                                             i * 384);
    }
    { float* t = cur; cur = alt; alt = t; }
    // mu = vectors @ mu_W                     [6144,128]x[128,256]
    gemm_wmma<128, 0, 0, 0><<<((BN * 3) / 16) * (256 / 64), 32, 0, stream>>>(
        cur, pmuW + (size_t)i * 128 * 256, nullptr, mu, 256, nullptr);
    post_kernel<<<BN, 128, 0, stream>>>(mu, scalars, ctx, vu);
    // t1 = silu(ctx @ aW1 + ab1)              [2048,256]x[256,128]
    gemm_wmma<256, 1, 0, 0><<<(BN / 16) * (128 / 64), 32, 0, stream>>>(
        ctx, paW1 + (size_t)i * 256 * 128, ab1 + i * 128, h1, 128, nullptr);
    // h2 = t1 @ aW2 + ab2                     [2048,128]x[128,384]
    gemm_wmma<128, 0, 0, 0><<<(BN / 16) * (384 / 64), 32, 0, stream>>>(
        h1, paW2 + (size_t)i * 128 * 384, ab2 + i * 384, hbuf, 384, nullptr);
    update_kernel<<<BN, 128, 0, stream>>>(hbuf, mu, vu, scalars, cur);
  }
}
